// FSSM_model_24292335026640
// MI455X (gfx1250) — matrix operand order repaired and checked by reading, not compiled, not run on hardware
//
#include <hip/hip_runtime.h>

// ============================================================================
// FSSM DEQ model on MI455X (gfx1250).
//
// Working set < 192MB => L2-resident, compute bound. All heavy math
// (rfft/irfft as DFT GEMMs, kernel spectrum, CH-MLPs, L-MLP head) runs on
// matrix cores via v_wmma_f32_16x16x32_bf16 with a bf16 hi/lo split
// (3 WMMAs per product => ~f32 accuracy at bf16 matrix rates).
// Each wave computes a 32x32 macro-tile: 12 WMMAs per k-step vs 8 aligned
// b128 A loads (B: 64 2-byte loads). DPLR Cauchy kernel + length-1024 iDFT are params-only VALU.
// Anderson acceleration uses deterministic LDS-tree reductions (no atomics).
// ============================================================================

typedef unsigned short ush;
typedef __attribute__((ext_vector_type(16))) __bf16 v16bf;
typedef __attribute__((ext_vector_type(8)))  float  v8f;

static constexpr int  kB   = 4;
static constexpr int  kL   = 1024;
static constexpr int  kCH  = 384;
static constexpr int  kN   = 64;
static constexpr int  kNJ  = kB * kCH;        // 1536 cols (b,c)
static constexpr int  kMF  = 1056;            // 1025 rfft rows -> 33*32
static constexpr int  kKF  = 1056;            // 1025 freq K    -> 33*32
static constexpr int  kMR  = kB * kL;         // 4096 rows (b,l)
static constexpr long kD   = (long)kL * kCH;  // 393216 per-batch DEQ dim
static constexpr long kTOT = (long)kB * kD;   // 1572864
static constexpr float kPI = 3.14159265358979323846f;
static constexpr float kANDLAM = 0.01f;

// ---------------------------------------------------------------- helpers ---
__device__ __forceinline__ ush f2bf(float x) {
  unsigned u = __float_as_uint(x);
  u += 0x7fffu + ((u >> 16) & 1u);
  return (ush)(u >> 16);
}
__device__ __forceinline__ float bf2f(ush h) {
  return __uint_as_float(((unsigned)h) << 16);
}
__device__ __forceinline__ void split2(float x, ush& h, ush& l) {
  h = f2bf(x);
  l = f2bf(x - bf2f(h));
}
__device__ __forceinline__ float gelu_t(float x) {
  const float c = 0.7978845608028654f;   // sqrt(2/pi), jax tanh-approx gelu
  return 0.5f * x * (1.0f + tanhf(c * (x + 0.044715f * x * x * x)));
}

struct cpx { float re, im; };
__device__ __forceinline__ cpx cadd(cpx a, cpx b) { return {a.re + b.re, a.im + b.im}; }
__device__ __forceinline__ cpx csub(cpx a, cpx b) { return {a.re - b.re, a.im - b.im}; }
__device__ __forceinline__ cpx cmulc(cpx a, cpx b) {
  return {a.re * b.re - a.im * b.im, a.re * b.im + a.im * b.re};
}
__device__ __forceinline__ cpx crcp(cpx a) {
  float d = a.re * a.re + a.im * a.im;
  float id = 1.0f / d;
  return {a.re * id, -a.im * id};
}
__device__ __forceinline__ cpx cdivc(cpx a, cpx b) { return cmulc(a, crcp(b)); }

// -------------------------------------------------------- DFT table build ---
// Forward rfft(n=2048) tables: Wc[f,t]=cos(2pi f t/2048), Ws[f,t]=-sin(..)
// rows f in [0,1056), zero for f>1024.  Stored as bf16 hi/lo split.
__global__ void fssm_build_fwd(ush* Wch, ush* Wcl, ush* Wsh, ush* Wsl) {
  long i = blockIdx.x * (long)blockDim.x + threadIdx.x;
  long total = (long)kMF * kL;
  if (i >= total) return;
  int f = (int)(i / kL);
  int t = (int)(i % kL);
  float cv = 0.f, sv = 0.f;
  if (f <= kL) {
    int p = (f * t) & (2 * kL - 1);           // phase index mod 2048 (exact)
    float th = (float)p * (kPI / (float)kL);  // 2*pi*p/2048
    float s, c; __sincosf(th, &s, &c);
    cv = c; sv = -s;
  }
  ush h, l;
  split2(cv, h, l); Wch[i] = h; Wcl[i] = l;
  split2(sv, h, l); Wsh[i] = h; Wsl[i] = l;
}

// Inverse irfft(n=2048) (first 1024 samples): Ci[t,f]=w_f cos(.)/2048,
// Si[t,f]=-w_f sin(.)/2048 with w_0=w_1024=1 else 2; cols f>1024 zero.
__global__ void fssm_build_inv(ush* Cih, ush* Cil, ush* Sih, ush* Sil) {
  long i = blockIdx.x * (long)blockDim.x + threadIdx.x;
  long total = (long)kL * kKF;
  if (i >= total) return;
  int t = (int)(i / kKF);
  int f = (int)(i % kKF);
  float cv = 0.f, sv = 0.f;
  if (f <= kL) {
    float w = (f == 0 || f == kL) ? 1.f : 2.f;
    int p = (f * t) & (2 * kL - 1);
    float th = (float)p * (kPI / (float)kL);
    float s, c; __sincosf(th, &s, &c);
    float sc = w / (float)(2 * kL);
    cv = c * sc; sv = -s * sc;
  }
  ush h, l;
  split2(cv, h, l); Cih[i] = h; Cil[i] = l;
  split2(sv, h, l); Sih[i] = h; Sil[i] = l;
}

// ----------------------------------------------------- DPLR kernel (S4) -----
// at[c,f] = (2/om1)*(k00 - k01*k10/(1+k11)) via Cauchy dots over N=64 modes.
__global__ void fssm_dplr_at(const float* B_re, const float* B_im,
                             const float* C_re, const float* C_im,
                             const float* P_re, const float* P_im,
                             const float* Q_re, const float* Q_im,
                             const float* dg_re, const float* dg_im,
                             const float* stepP,
                             float* at_re, float* at_im) {
  int f = blockIdx.x * blockDim.x + threadIdx.x;
  int c = blockIdx.y;
  float step = fmaxf(stepP[0], 1e-6f);
  float ang = -2.0f * kPI * (float)f / (float)kL;
  cpx Om; __sincosf(ang, &Om.im, &Om.re);
  cpx om1{1.f + Om.re, Om.im};
  if (hypotf(om1.re, om1.im) < 1e-7f) { om1.re = 1e-7f; om1.im = 0.f; }
  cpx one_m{1.f - Om.re, -Om.im};
  cpx g = cmulc(cpx{2.0f / step, 0.f}, cdivc(one_m, om1));
  cpx k00{0, 0}, k01{0, 0}, k10{0, 0}, k11{0, 0};
  for (int n = 0; n < kN; ++n) {
    cpx d{dg_re[n], dg_im[n]};
    cpx inv = crcp(cpx{g.re - d.re, g.im - d.im});
    cpx Cv{C_re[c * kN + n], C_im[c * kN + n]};
    cpx Bv{B_re[n], B_im[n]};
    cpx Pv{P_re[n], P_im[n]};
    cpx Qv{Q_re[n], Q_im[n]};
    k00 = cadd(k00, cmulc(cmulc(Cv, Bv), inv));
    k01 = cadd(k01, cmulc(cmulc(Cv, Pv), inv));
    k10 = cadd(k10, cmulc(inv, cmulc(Qv, Bv)));
    k11 = cadd(k11, cmulc(inv, cmulc(Qv, Pv)));
  }
  cpx corr = cmulc(k01, cdivc(k10, cpx{1.f + k11.re, k11.im}));
  cpx at = cmulc(cdivc(cpx{2.f, 0.f}, om1), csub(k00, corr));
  at_re[c * kL + f] = at.re;
  at_im[c * kL + f] = at.im;
}

// K[c,l] = (1/L) Re sum_f at[c,f] e^{+2pi i f l / L}, rotation recurrence
// with periodic exact resync.
__global__ void fssm_idft_K(const float* at_re, const float* at_im, float* Kt) {
  int l = blockIdx.x * blockDim.x + threadIdx.x;
  int c = blockIdx.y;
  float phi = 2.0f * kPI * (float)l / (float)kL;
  float cp, sp; __sincosf(phi, &sp, &cp);
  float cc = 1.f, ss = 0.f, acc = 0.f;
  const float* ar = at_re + (long)c * kL;
  const float* ai = at_im + (long)c * kL;
  for (int f = 0; f < kL; ++f) {
    if ((f & 63) == 0) {
      int p = (l * f) & (kL - 1);
      float th = (2.0f * kPI / (float)kL) * (float)p;
      __sincosf(th, &ss, &cc);
    }
    acc += ar[f] * cc - ai[f] * ss;
    float c2 = cc * cp - ss * sp;
    ss = ss * cp + cc * sp;
    cc = c2;
  }
  Kt[c * kL + l] = acc / (float)kL;
}

// ------------------------------------------------------------ pack kernels --
// dst[r,c] = src[r*rs + c*cs] (zero-pad outside R x C), bf16 hi/lo split.
__global__ void fssm_pack_strided(ush* dh, ush* dl, const float* src,
                                  int R, int C, long rs, long cs,
                                  int dR, int dC) {
  long i = blockIdx.x * (long)blockDim.x + threadIdx.x;
  long total = (long)dR * dC;
  if (i >= total) return;
  int r = (int)(i / dC), c = (int)(i % dC);
  float v = (r < R && c < C) ? src[(long)r * rs + (long)c * cs] : 0.f;
  ush h, l; split2(v, h, l);
  dh[i] = h; dl[i] = l;
}

// u = src - gamma*feed in (b,l,c), plus transposed pack to (t, j=(b,c)).
__global__ void fssm_pack_u(const float* src, const float* feed,
                            const float* gammaP,
                            float* uf32, ush* Uph, ush* Upl) {
  long i = blockIdx.x * (long)blockDim.x + threadIdx.x;
  long total = (long)kL * kNJ;
  if (i >= total) return;
  int t = (int)(i / kNJ), j = (int)(i % kNJ);
  int b = j / kCH, c = j % kCH;
  long si = ((long)b * kL + t) * kCH + c;
  float v = src[si];
  if (feed) v -= gammaP[0] * feed[si];
  uf32[si] = v;
  ush h, l; split2(v, h, l);
  Uph[i] = h; Upl[i] = l;
}

// H = U * Khat (complex, per channel), output bf16 hi/lo; rows f>1024 zeroed.
__global__ void fssm_cmul(const float* Ure, const float* Uim,
                          const float* Khre, const float* Khim,
                          ush* Hreh, ush* Hrel, ush* Himh, ush* Himl) {
  long i = blockIdx.x * (long)blockDim.x + threadIdx.x;
  long total = (long)kKF * kNJ;
  if (i >= total) return;
  int f = (int)(i / kNJ), j = (int)(i % kNJ), c = j % kCH;
  float hre = 0.f, him = 0.f;
  if (f <= kL) {
    float ur = Ure[(long)f * kNJ + j], ui = Uim[(long)f * kNJ + j];
    float kr = Khre[(long)f * kCH + c], ki = Khim[(long)f * kCH + c];
    hre = ur * kr - ui * ki;
    him = ur * ki + ui * kr;
  }
  ush h, l;
  split2(hre, h, l); Hreh[i] = h; Hrel[i] = l;
  split2(him, h, l); Himh[i] = h; Himl[i] = l;
}

// y[b,l,c] = gelu(h[t=l, j=(b,c)] + D * u[b,l,c])
__global__ void fssm_hepi(const float* hbuf, const float* uf32,
                          const float* Dp, float* y) {
  long i = blockIdx.x * (long)blockDim.x + threadIdx.x;
  if (i >= kTOT) return;
  int m = (int)(i / kCH);
  int c = (int)(i % kCH);
  int b = m / kL, t = m % kL;
  long hj = (long)t * kNJ + (long)b * kCH + c;
  y[i] = gelu_t(hbuf[hj] + Dp[0] * uf32[i]);
}

// ------------------------------------------------------------- WMMA GEMM ----
// C(MxN,f32) = split-bf16 A(MxK) @ split-bf16 B(KxN), 3 wmma per product.
// One wave = one 32x32 macro-tile (2x2 WMMA tiles, 4 accumulators, 12 WMMAs
// per k-step).  Block = (32,4) => 4 waves span 128 columns.
// Grid = (N/128, M/32, batch).  All dims are padded multiples, so EXEC is
// all-ones around every WMMA.  A fragment loads are aligned b128; B is loaded per 16-bit element.
__global__ __launch_bounds__(128)
void fssm_wmma_gemm(const ush* __restrict__ Ah, const ush* __restrict__ Al,
                    const ush* __restrict__ Bh, const ush* __restrict__ Bl,
                    float* __restrict__ C,
                    int K, int lda, int ldb, int ldc,
                    long sA, long sB, long sC,
                    const float* __restrict__ bias, int biasMode, int act,
                    int accum) {
  union V16 { v16bf v; ush s[16]; uint4 q[2]; };
  const int lane = threadIdx.x;
  const int tn = (blockIdx.x * 4 + threadIdx.y) * 32;  // 32-wide col base
  const int tm = blockIdx.y * 32;                      // 32-tall row base
  const long zb = blockIdx.z;
  Ah += zb * sA; Al += zb * sA;
  Bh += zb * sB; Bl += zb * sB;
  C  += zb * sC;

  const int m  = lane & 15;   // row-in-tile (A) / col-in-tile (C)
  const int hs = lane >> 4;   // half select
  const long arow0 = (long)(tm + m) * lda;        // M-tile 0
  const long arow1 = (long)(tm + 16 + m) * lda;   // M-tile 1

  v8f acc00 = {0.f,0.f,0.f,0.f,0.f,0.f,0.f,0.f};
  v8f acc01 = acc00, acc10 = acc00, acc11 = acc00;

  for (int kk = 0; kk < K; kk += 32) {
    V16 a0h, a0l, a1h, a1l, b0h, b0l, b1h, b1l;
    // A 16x32 fragment, ISA layout: lane m holds K = [8hs..8hs+7] (s0..7)
    // and [16+8hs..16+8hs+7] (s8..15); both 16B groups are 16B-aligned.
    const long ka = kk + 8 * hs;
    a0h.q[0] = *reinterpret_cast<const uint4*>(Ah + arow0 + ka);
    a0h.q[1] = *reinterpret_cast<const uint4*>(Ah + arow0 + ka + 16);
    a0l.q[0] = *reinterpret_cast<const uint4*>(Al + arow0 + ka);
    a0l.q[1] = *reinterpret_cast<const uint4*>(Al + arow0 + ka + 16);
    a1h.q[0] = *reinterpret_cast<const uint4*>(Ah + arow1 + ka);
    a1h.q[1] = *reinterpret_cast<const uint4*>(Ah + arow1 + ka + 16);
    a1l.q[0] = *reinterpret_cast<const uint4*>(Al + arow1 + ka);
    a1l.q[1] = *reinterpret_cast<const uint4*>(Al + arow1 + ka + 16);
    // B 32x16 fragments: lane holds col m, K in the same order as A.
    const long brow = (long)(kk + 8 * hs) * ldb + tn + m;
    for (int s = 0; s < 16; ++s) {
      const long bo = brow + (long)((s & 7) + ((s >> 3) << 4)) * ldb;
      b0h.s[s] = Bh[bo]; b0l.s[s] = Bl[bo];
      b1h.s[s] = Bh[bo + 16]; b1l.s[s] = Bl[bo + 16];
    }
    if (kk + 32 < K) {
      __builtin_prefetch(Bh + brow + (long)32 * ldb, 0, 1);
      __builtin_prefetch(Ah + arow0 + ka + 32, 0, 1);
    }
    // 12 WMMAs: hi*hi + hi*lo + lo*hi per output tile.
    acc00 = __builtin_amdgcn_wmma_f32_16x16x32_bf16(false, a0h.v, false, b0h.v, (short)0, acc00, false, false);
    acc00 = __builtin_amdgcn_wmma_f32_16x16x32_bf16(false, a0h.v, false, b0l.v, (short)0, acc00, false, false);
    acc00 = __builtin_amdgcn_wmma_f32_16x16x32_bf16(false, a0l.v, false, b0h.v, (short)0, acc00, false, false);
    acc01 = __builtin_amdgcn_wmma_f32_16x16x32_bf16(false, a0h.v, false, b1h.v, (short)0, acc01, false, false);
    acc01 = __builtin_amdgcn_wmma_f32_16x16x32_bf16(false, a0h.v, false, b1l.v, (short)0, acc01, false, false);
    acc01 = __builtin_amdgcn_wmma_f32_16x16x32_bf16(false, a0l.v, false, b1h.v, (short)0, acc01, false, false);
    acc10 = __builtin_amdgcn_wmma_f32_16x16x32_bf16(false, a1h.v, false, b0h.v, (short)0, acc10, false, false);
    acc10 = __builtin_amdgcn_wmma_f32_16x16x32_bf16(false, a1h.v, false, b0l.v, (short)0, acc10, false, false);
    acc10 = __builtin_amdgcn_wmma_f32_16x16x32_bf16(false, a1l.v, false, b0h.v, (short)0, acc10, false, false);
    acc11 = __builtin_amdgcn_wmma_f32_16x16x32_bf16(false, a1h.v, false, b1h.v, (short)0, acc11, false, false);
    acc11 = __builtin_amdgcn_wmma_f32_16x16x32_bf16(false, a1h.v, false, b1l.v, (short)0, acc11, false, false);
    acc11 = __builtin_amdgcn_wmma_f32_16x16x32_bf16(false, a1l.v, false, b1h.v, (short)0, acc11, false, false);
  }
  // C/D layout: VGPR r -> row r+8*hs, col lane&15 (per 16x16 tile).
#pragma unroll
  for (int ti = 0; ti < 2; ++ti) {
#pragma unroll
    for (int tj = 0; tj < 2; ++tj) {
      const v8f& acc = ti ? (tj ? acc11 : acc10) : (tj ? acc01 : acc00);
#pragma unroll
      for (int r = 0; r < 8; ++r) {
        int grow = tm + ti * 16 + r + 8 * hs;
        int gcol = tn + tj * 16 + m;
        long ci = (long)grow * ldc + gcol;
        float v = acc[r];
        if (accum) v += C[ci];
        if (biasMode == 1) v += bias[gcol];
        else if (biasMode == 2) v += bias[grow];
        if (act == 1) v = gelu_t(v);
        else if (act == 2) v = tanhf(v);
        C[ci] = v;
      }
    }
  }
}

// ----------------------------------------------------- Anderson machinery ---
struct HistPtrs { const float* hx[5]; const float* hf[5]; };

__global__ void fssm_axupd(float* feed, const float* fb) {
  long i = blockIdx.x * (long)blockDim.x + threadIdx.x;
  if (i >= kTOT) return;
  feed[i] = 0.3f * feed[i] + 0.7f * fb[i];
}

__global__ void fssm_dots(HistPtrs hp, int k, float* GTG, float* GTF) {
  __shared__ float red[256];
  int p = blockIdx.x, b = blockIdx.y, tid = threadIdx.x;
  int kk1 = k - 1;
  int ii, jj;
  if (p < kk1 * kk1) { ii = p / kk1; jj = p % kk1; }
  else               { ii = p - kk1 * kk1; jj = -1; }
  long base = (long)b * kD;
  float sum = 0.f;
  for (long d = tid; d < kD; d += 256) {
    long e = base + d;
    float gi = (hp.hf[ii + 1][e] - hp.hx[ii + 1][e]) -
               (hp.hf[ii][e]     - hp.hx[ii][e]);
    float ot;
    if (jj >= 0)
      ot = (hp.hf[jj + 1][e] - hp.hx[jj + 1][e]) -
           (hp.hf[jj][e]     - hp.hx[jj][e]);
    else
      ot = hp.hf[k - 1][e] - hp.hx[k - 1][e];
    sum += gi * ot;
  }
  red[tid] = sum; __syncthreads();
  for (int s = 128; s > 0; s >>= 1) {
    if (tid < s) red[tid] += red[tid + s];
    __syncthreads();
  }
  if (tid == 0) {
    if (jj >= 0) GTG[b * 16 + ii * 4 + jj] = red[0];
    else         GTF[b * 4 + ii] = red[0];
  }
}

__global__ void fssm_solve(const float* GTG, const float* GTF, int kk1,
                           float lam, float* alpha, float* asum) {
  int b = threadIdx.x;
  if (b >= kB) return;
  float A[4][4], rhs[4], sol[4];
  for (int i = 0; i < kk1; ++i) {
    for (int j = 0; j < kk1; ++j)
      A[i][j] = GTG[b * 16 + i * 4 + j] + ((i == j) ? lam : 0.f);
    rhs[i] = -GTF[b * 4 + i];
  }
  for (int col = 0; col < kk1; ++col) {
    float ip = 1.f / A[col][col];
    for (int row = col + 1; row < kk1; ++row) {
      float f = A[row][col] * ip;
      for (int c2 = col; c2 < kk1; ++c2) A[row][c2] -= f * A[col][c2];
      rhs[row] -= f * rhs[col];
    }
  }
  float s = 0.f;
  for (int row = kk1 - 1; row >= 0; --row) {
    float v = rhs[row];
    for (int c2 = row + 1; c2 < kk1; ++c2) v -= A[row][c2] * sol[c2];
    sol[row] = v / A[row][row];
  }
  for (int i = 0; i < kk1; ++i) { alpha[b * 4 + i] = sol[i]; s += sol[i]; }
  for (int i = kk1; i < 4; ++i) alpha[b * 4 + i] = 0.f;
  asum[b] = s;
}

__global__ void fssm_combine(HistPtrs hp, int kk1, const float* fb,
                             const float* alpha, const float* asum,
                             float* feed) {
  long i = blockIdx.x * (long)blockDim.x + threadIdx.x;
  if (i >= kTOT) return;
  int b = (int)(i / kD);
  float v = (1.f - asum[b]) * fb[i];
  for (int t = 0; t < kk1; ++t) v += alpha[b * 4 + t] * hp.hf[t][i];
  feed[i] = v;
}

// ============================================================================
// Host orchestration
// ============================================================================
extern "C" void kernel_launch(void* const* d_in, const int* in_sizes, int n_in,
                              void* d_out, int out_size, void* d_ws,
                              size_t ws_size, hipStream_t stream) {
  (void)in_sizes; (void)n_in; (void)out_size; (void)ws_size;

  // ---- input leaves (insertion order of setup_inputs pytree) --------------
  auto F = [&](int i) { return (const float*)d_in[i]; };
  const float* x      = F(0);
  const float* gammaP = F(1);
  struct Blk {
    const float *B_re, *B_im, *C_re, *C_im, *P_re, *P_im, *Q_re, *Q_im;
    const float *dg_re, *dg_im, *step, *D, *fc_w, *fc_b;
  } blk[3];
  const int base[3] = {2, 16, 30};
  for (int q = 0; q < 3; ++q) {
    int o = base[q];
    blk[q] = {F(o + 0), F(o + 1), F(o + 2), F(o + 3), F(o + 4), F(o + 5),
              F(o + 6), F(o + 7), F(o + 8), F(o + 9), F(o + 10), F(o + 11),
              F(o + 12), F(o + 13)};
  }
  const float* fc1_w = F(44);
  const float* fc1_b = F(45);
  const float* fc2_w = F(46);
  const float* fc2_b = F(47);

  // ---- workspace bump allocator -------------------------------------------
  char* wsb = (char*)d_ws;
  size_t off = 0;
  auto alloc = [&](size_t bytes) -> void* {
    void* p = wsb + off;
    off = (off + bytes + 255) & ~((size_t)255);
    return p;
  };
  const size_t fwdN = (size_t)kMF * kL;
  const size_t invN = (size_t)kL * kKF;
  ush* Wc_h = (ush*)alloc(fwdN * 2);  ush* Wc_l = (ush*)alloc(fwdN * 2);
  ush* Ws_h = (ush*)alloc(fwdN * 2);  ush* Ws_l = (ush*)alloc(fwdN * 2);
  ush* Ci_h = (ush*)alloc(invN * 2);  ush* Ci_l = (ush*)alloc(invN * 2);
  ush* Si_h = (ush*)alloc(invN * 2);  ush* Si_l = (ush*)alloc(invN * 2);
  float *Khre[3], *Khim[3];
  for (int q = 0; q < 3; ++q) {
    Khre[q] = (float*)alloc((size_t)kKF * kCH * 4);
    Khim[q] = (float*)alloc((size_t)kKF * kCH * 4);
  }
  float* at_re = (float*)alloc((size_t)kCH * kL * 4);
  float* at_im = (float*)alloc((size_t)kCH * kL * 4);
  float* Ktime = (float*)alloc((size_t)kCH * kL * 4);
  ush* Ktp_h = (ush*)alloc((size_t)kL * kCH * 2);
  ush* Ktp_l = (ush*)alloc((size_t)kL * kCH * 2);
  ush *fcw_h[3], *fcw_l[3];
  for (int q = 0; q < 3; ++q) {
    fcw_h[q] = (ush*)alloc((size_t)kCH * kCH * 2);
    fcw_l[q] = (ush*)alloc((size_t)kCH * kCH * 2);
  }
  ush* fc2_h = (ush*)alloc((size_t)kCH * kCH * 2);
  ush* fc2_l = (ush*)alloc((size_t)kCH * kCH * 2);
  ush* fc1_h = (ush*)alloc((size_t)kL * kL * 2);
  ush* fc1_l = (ush*)alloc((size_t)kL * kL * 2);
  float* uf32 = (float*)alloc((size_t)kTOT * 4);
  ush* Up_h = (ush*)alloc((size_t)kL * kNJ * 2);
  ush* Up_l = (ush*)alloc((size_t)kL * kNJ * 2);
  float* Ure = (float*)alloc((size_t)kMF * kNJ * 4);
  float* Uim = (float*)alloc((size_t)kMF * kNJ * 4);
  ush* Hre_h = (ush*)alloc((size_t)kKF * kNJ * 2);
  ush* Hre_l = (ush*)alloc((size_t)kKF * kNJ * 2);
  ush* Him_h = (ush*)alloc((size_t)kKF * kNJ * 2);
  ush* Him_l = (ush*)alloc((size_t)kKF * kNJ * 2);
  float* hbuf = (float*)alloc((size_t)kL * kNJ * 4);
  float* ycur = (float*)alloc((size_t)kTOT * 4);
  float* ynext = (float*)alloc((size_t)kTOT * 4);
  float* ytmp = (float*)alloc((size_t)kTOT * 4);
  ush* Ap_h = (ush*)alloc((size_t)kMR * kCH * 2);
  ush* Ap_l = (ush*)alloc((size_t)kMR * kCH * 2);
  float* Fbuf = (float*)alloc((size_t)kTOT * 4);
  float* FB = (float*)alloc((size_t)kTOT * 4);
  float* feed = (float*)alloc((size_t)kTOT * 4);
  float* HX[5];
  float* HF[5];
  for (int q = 0; q < 5; ++q) HX[q] = (float*)alloc((size_t)kTOT * 4);
  for (int q = 0; q < 5; ++q) HF[q] = (float*)alloc((size_t)kTOT * 4);
  float* GTG = (float*)alloc(kB * 16 * 4);
  float* GTF = (float*)alloc(kB * 4 * 4);
  float* alphaB = (float*)alloc(kB * 4 * 4);
  float* asumB = (float*)alloc(kB * 4);

  auto g1 = [](long n) { return dim3((unsigned)((n + 255) / 256)); };
  auto gemm = [&](const ush* Ah, const ush* Al, const ush* Bh, const ush* Bl,
                  float* C, int M, int N, int K, int lda, int ldb, int ldc,
                  long sA, long sB, long sC, int batch, const float* bias,
                  int bm, int act, int acc) {
    dim3 grd(N / 128, M / 32, batch), blkd(32, 4, 1);
    fssm_wmma_gemm<<<grd, blkd, 0, stream>>>(Ah, Al, Bh, Bl, C, K, lda, ldb,
                                             ldc, sA, sB, sC, bias, bm, act,
                                             acc);
  };

  // ---- per-call setup (deterministic, recomputed every call) --------------
  fssm_build_fwd<<<g1((long)kMF * kL), 256, 0, stream>>>(Wc_h, Wc_l, Ws_h, Ws_l);
  fssm_build_inv<<<g1((long)kL * kKF), 256, 0, stream>>>(Ci_h, Ci_l, Si_h, Si_l);
  for (int q = 0; q < 3; ++q) {
    fssm_dplr_at<<<dim3(kL / 128, kCH), 128, 0, stream>>>(
        blk[q].B_re, blk[q].B_im, blk[q].C_re, blk[q].C_im, blk[q].P_re,
        blk[q].P_im, blk[q].Q_re, blk[q].Q_im, blk[q].dg_re, blk[q].dg_im,
        blk[q].step, at_re, at_im);
    fssm_idft_K<<<dim3(kL / 128, kCH), 128, 0, stream>>>(at_re, at_im, Ktime);
    // Kt^T pack: dst[l,c] = Ktime[c*kL + l]
    fssm_pack_strided<<<g1((long)kL * kCH), 256, 0, stream>>>(
        Ktp_h, Ktp_l, Ktime, kL, kCH, 1, kL, kL, kCH);
    gemm(Wc_h, Wc_l, Ktp_h, Ktp_l, Khre[q], kMF, kCH, kL, kL, kCH, kCH,
         0, 0, 0, 1, nullptr, 0, 0, 0);
    gemm(Ws_h, Ws_l, Ktp_h, Ktp_l, Khim[q], kMF, kCH, kL, kL, kCH, kCH,
         0, 0, 0, 1, nullptr, 0, 0, 0);
    // fc weight as B operand: dst[i,o] = fc_w[o,i]
    fssm_pack_strided<<<g1((long)kCH * kCH), 256, 0, stream>>>(
        fcw_h[q], fcw_l[q], blk[q].fc_w, kCH, kCH, 1, kCH, kCH, kCH);
  }
  fssm_pack_strided<<<g1((long)kCH * kCH), 256, 0, stream>>>(
      fc2_h, fc2_l, fc2_w, kCH, kCH, 1, kCH, kCH, kCH);
  fssm_pack_strided<<<g1((long)kL * kL), 256, 0, stream>>>(
      fc1_h, fc1_l, fc1_w, kL, kL, kL, 1, kL, kL);
  hipMemsetAsync(feed, 0, (size_t)kTOT * 4, stream);

  // ---- one full pass: input/mid/output blocks (+ optional DEQ head) -------
  auto passFn = [&](const float* feedP, bool withHead, float* yFinal) {
    fssm_pack_u<<<g1((long)kL * kNJ), 256, 0, stream>>>(x, feedP, gammaP,
                                                        uf32, Up_h, Up_l);
    for (int q = 0; q < 3; ++q) {
      gemm(Wc_h, Wc_l, Up_h, Up_l, Ure, kMF, kNJ, kL, kL, kNJ, kNJ,
           0, 0, 0, 1, nullptr, 0, 0, 0);
      gemm(Ws_h, Ws_l, Up_h, Up_l, Uim, kMF, kNJ, kL, kL, kNJ, kNJ,
           0, 0, 0, 1, nullptr, 0, 0, 0);
      fssm_cmul<<<g1((long)kKF * kNJ), 256, 0, stream>>>(
          Ure, Uim, Khre[q], Khim[q], Hre_h, Hre_l, Him_h, Him_l);
      gemm(Ci_h, Ci_l, Hre_h, Hre_l, hbuf, kL, kNJ, kKF, kKF, kNJ, kNJ,
           0, 0, 0, 1, nullptr, 0, 0, 0);
      gemm(Si_h, Si_l, Him_h, Him_l, hbuf, kL, kNJ, kKF, kKF, kNJ, kNJ,
           0, 0, 0, 1, nullptr, 0, 0, 1);
      fssm_hepi<<<g1(kTOT), 256, 0, stream>>>(hbuf, uf32, blk[q].D, ycur);
      fssm_pack_strided<<<g1(kTOT), 256, 0, stream>>>(
          Ap_h, Ap_l, ycur, kMR, kCH, kCH, 1, kMR, kCH);
      float* Cout = (q == 2) ? yFinal : ynext;
      gemm(Ap_h, Ap_l, fcw_h[q], fcw_l[q], Cout, kMR, kCH, kCH, kCH, kCH, kCH,
           0, 0, 0, 1, blk[q].fc_b, 1, 1, 0);
      if (q < 2)
        fssm_pack_u<<<g1((long)kL * kNJ), 256, 0, stream>>>(
            Cout, nullptr, nullptr, uf32, Up_h, Up_l);
    }
    if (withHead) {
      fssm_pack_strided<<<g1(kTOT), 256, 0, stream>>>(
          Ap_h, Ap_l, yFinal, kMR, kCH, kCH, 1, kMR, kCH);
      gemm(Ap_h, Ap_l, fc2_h, fc2_l, Fbuf, kMR, kCH, kCH, kCH, kCH, kCH,
           0, 0, 0, 1, fc2_b, 1, 0, 0);
      fssm_pack_strided<<<g1(kTOT), 256, 0, stream>>>(
          Ap_h, Ap_l, Fbuf, kMR, kCH, kCH, 1, kMR, kCH);
      // out[b,k,c] = tanh(fc1_w @ F_b + fc1_b[k]); batched over b
      gemm(fc1_h, fc1_l, Ap_h, Ap_l, FB, kL, kCH, kL, kL, kCH, kCH,
           0, (long)kL * kCH, (long)kL * kCH, kB, fc1_b, 2, 2, 0);
    }
  };

  // ---- DEQ loop with Anderson acceleration --------------------------------
  const int DEQ_ITERS = 20, AND_M = 5;
  for (int it = 0; it < DEQ_ITERS; ++it) {
    passFn(feed, true, ytmp);  // fb -> FB
    int slot = it % AND_M;
    hipMemcpyAsync(HX[slot], feed, (size_t)kTOT * 4,
                   hipMemcpyDeviceToDevice, stream);
    hipMemcpyAsync(HF[slot], FB, (size_t)kTOT * 4,
                   hipMemcpyDeviceToDevice, stream);
    int k = (it + 1 < AND_M) ? it + 1 : AND_M;
    if (k < 2) {
      fssm_axupd<<<g1(kTOT), 256, 0, stream>>>(feed, FB);
    } else {
      HistPtrs hp{};
      for (int j = 0; j < k; ++j) {
        int s = (it - (k - 1) + j) % AND_M;
        hp.hx[j] = HX[s];
        hp.hf[j] = HF[s];
      }
      int kk1 = k - 1;
      int npairs = kk1 * kk1 + kk1;
      fssm_dots<<<dim3(npairs, kB), 256, 0, stream>>>(hp, k, GTG, GTF);
      fssm_solve<<<1, kB, 0, stream>>>(GTG, GTF, kk1, kANDLAM, alphaB, asumB);
      fssm_combine<<<g1(kTOT), 256, 0, stream>>>(hp, kk1, FB, alphaB, asumB,
                                                 feed);
    }
  }
  // ---- final differentiable pass; output block fc writes d_out directly ---
  passFn(feed, false, (float*)d_out);
}
